// EpisodicMemory_89876485636308
// MI455X (gfx1250) — compile-verified, matrix operand order checked by reading, not run on hardware
//
#include <hip/hip_runtime.h>
#include <math.h>

#define BS_ 8
#define B_  4
#define M_  256
#define D_  256
#define N_  2048
#define TN  128          // rows of y per workgroup (8 waves x 16 rows)
#define NT  (N_/TN)      // 16
#define NSTEPS 2

typedef __attribute__((ext_vector_type(16))) __bf16 v16bf;
typedef __attribute__((ext_vector_type(8)))  float  v8f;

// ---------------- LDS layout ----------------
// KF: em_K as 128 A-layout bf16 fragments (frag(kc=d>>5, mt=m>>4)) : 128 KB
// VF: em_V^T as 128 A-layout bf16 fragments (frag(mc=m>>5, dt=d>>4)): 128 KB
// SB: per-m additive mask bias (0 or -1e30)                         : 1 KB
#define LDS_KF 0
#define LDS_VF 131072
#define LDS_SB 262144
#define LDS_TOTAL (262144 + 1024)

__device__ __forceinline__ unsigned short f2bf(float f) {
  union { float f; unsigned u; } x; x.f = f;
  unsigned u = x.u;
  u += 0x7FFFu + ((u >> 16) & 1u);   // round-to-nearest-even
  return (unsigned short)(u >> 16);
}
__device__ __forceinline__ unsigned pack2(float lo, float hi) {
  return (unsigned)f2bf(lo) | ((unsigned)f2bf(hi) << 16);
}

// Byte offset inside a 1KB 16(M/N)x32(K) bf16 fragment for element (k, n).
// Layout (A-operand, per CDNA5 ISA 7.12.2): lane = ((k&15)>>3)*16 + n,
// slot within lane's 16 bf16 = v*2+p with v = (k>>4)*4 + ((k&7)>>1), p = k&1.
__device__ __forceinline__ int fragOff(int k, int n) {
  int hi   = k >> 4;
  int kk   = k & 15;
  int half = kk >> 3;
  int r    = kk & 7;
  int v    = hi * 4 + (r >> 1);
  int p    = r & 1;
  int lane = half * 16 + n;
  return (lane * 16 + v * 2 + p) * 2;
}

// out = -4 * seed   (so that later  out += y_b  for b=0..3 yields sum_b (y_b - s))
__global__ void ep_init_out(const float* __restrict__ seed, float* __restrict__ out, int n) {
  int i = blockIdx.x * blockDim.x + threadIdx.x;
  if (i < n) out[i] = -4.0f * seed[i];
}

__global__ __launch_bounds__(256, 1)
void ep_mem_kernel(const float* __restrict__ seed, const float* __restrict__ emK,
                   const float* __restrict__ emV,  const float* __restrict__ emS,
                   const float* __restrict__ gAlpha, const float* __restrict__ gBias,
                   const float* __restrict__ rTau,  float* __restrict__ out)
{
  extern __shared__ unsigned char smem[];
  const int tid  = threadIdx.x;
  const int lane = tid & 31;
  const int wave = tid >> 5;
  const int tile = blockIdx.x;   // 0..15
  const int b    = blockIdx.y;   // 0..3
  const int bs   = blockIdx.z;   // 0..7

  // per-slot scalars
  const float rt     = rTau[b];
  const float tau    = log1pf(__expf(rt)) + 0.1f;
  const float invTau = 1.0f / tau;
  const float alpha  = gAlpha[b];
  const float gbias  = gBias[b];

  // ---------------- stage K and V into LDS (f32 -> bf16, WMMA A-fragment swizzle) ----
  {
    const size_t base = (size_t)(bs * B_ + b) * M_ * D_;
    const float4* K4 = (const float4*)(emK + base);
    const float4* V4 = (const float4*)(emV + base);
    for (int i = tid; i < M_ * D_ / 4; i += 256) {
      int m  = i >> 6;             // row (m), 64 float4 per row
      int d0 = (i & 63) << 2;      // starting d, multiple of 4
      // K: scores^T GEMM uses A = K tile (M-dim = m_local, K-dim = d_local)
      float4 f = K4[i];
      unsigned char* kb = smem + LDS_KF + (((d0 >> 5) * 16 + (m >> 4)) << 10);
      int nk = m & 15;
      *(unsigned*)(kb + fragOff(d0 & 31, nk))       = pack2(f.x, f.y);
      *(unsigned*)(kb + fragOff((d0 + 2) & 31, nk)) = pack2(f.z, f.w);
      // V: delta^T GEMM uses A = V^T tile (M-dim = d_local, K-dim = m_local)
      float4 g = V4[i];
      int kv = m & 31;
      unsigned char* vb = smem + LDS_VF + (((m >> 5) * 16 + (d0 >> 4)) << 10);
      *(unsigned short*)(vb + fragOff(kv, (d0    ) & 15)) = f2bf(g.x);
      *(unsigned short*)(vb + fragOff(kv, (d0 + 1) & 15)) = f2bf(g.y);
      *(unsigned short*)(vb + fragOff(kv, (d0 + 2) & 15)) = f2bf(g.z);
      *(unsigned short*)(vb + fragOff(kv, (d0 + 3) & 15)) = f2bf(g.w);
    }
    float* sbw = (float*)(smem + LDS_SB);
    if (tid < M_) {
      float sv = emS[(size_t)(bs * B_ + b) * M_ + tid];
      sbw[tid] = (sv > 0.0f) ? 0.0f : -1e30f;   // additive softmax mask
    }
  }
  __syncthreads();

  const int   half   = lane >> 4;
  const int   col    = lane & 15;
  const int   myRow  = tile * TN + wave * 16 + col;     // n index owned by this lane
  const float* sbias = (const float*)(smem + LDS_SB);

  // y^T in C-fragment layout: yT[t][r] = y[myRow][t*16 + r + 8*half]
  float yT[16][8];
  {
    const float* srow = seed + ((size_t)bs * N_ + myRow) * D_;
    #pragma unroll
    for (int t = 0; t < 16; ++t) {
      const float4* p4 = (const float4*)(srow + t * 16 + 8 * half);
      float4 a = p4[0], c = p4[1];
      yT[t][0] = a.x; yT[t][1] = a.y; yT[t][2] = a.z; yT[t][3] = a.w;
      yT[t][4] = c.x; yT[t][5] = c.y; yT[t][6] = c.z; yT[t][7] = c.w;
    }
  }

  const v8f vzero = {};

  #pragma unroll 1
  for (int step = 0; step < NSTEPS; ++step) {
    // ---- bf16 B-operand fragments of y^T (lane-local pack, one per 32-wide d-chunk)
    v16bf By[8];
    #pragma unroll
    for (int kc = 0; kc < 8; ++kc) {
      union { v16bf v; unsigned u[8]; } uu;
      #pragma unroll
      for (int v = 0; v < 8; ++v) {
        int t = 2 * kc + (v >> 2);
        int r = (v & 3) * 2;
        uu.u[v] = pack2(yT[t][r], yT[t][r + 1]);
      }
      By[kc] = uu.v;
    }

    v8f dacc[16];                          // delta^T accumulators (unnormalized)
    #pragma unroll
    for (int t = 0; t < 16; ++t) dacc[t] = vzero;
    float rowMax = -3.0e38f;
    float rowSum = 0.0f;

    #pragma unroll 1
    for (int mc = 0; mc < 8; ++mc) {       // online softmax over 32-wide m-chunks
      // GEMM1: scores^T chunk = K(chunk rows) x y^T
      v8f sc0 = vzero, sc1 = vzero;
      #pragma unroll
      for (int kc = 0; kc < 8; ++kc) {
        const unsigned char* ka = smem + LDS_KF + lane * 32;
        v16bf a0 = *(const v16bf*)(ka + ((kc * 16 + 2 * mc    ) << 10));
        v16bf a1 = *(const v16bf*)(ka + ((kc * 16 + 2 * mc + 1) << 10));
        sc0 = __builtin_amdgcn_wmma_f32_16x16x32_bf16(false, a0, false, By[kc],
                                                      (short)0, sc0, false, false);
        sc1 = __builtin_amdgcn_wmma_f32_16x16x32_bf16(false, a1, false, By[kc],
                                                      (short)0, sc1, false, false);
      }
      // scale + mask, chunk max
      float s0[8], s1[8];
      float cmax = -3.0e38f;
      #pragma unroll
      for (int r = 0; r < 8; ++r) {
        int m0 = (2 * mc    ) * 16 + r + 8 * half;
        int m1 = (2 * mc + 1) * 16 + r + 8 * half;
        s0[r] = sc0[r] * invTau + sbias[m0];
        s1[r] = sc1[r] * invTau + sbias[m1];
        cmax  = fmaxf(cmax, fmaxf(s0[r], s1[r]));
      }
      cmax = fmaxf(cmax, __shfl_xor(cmax, 16, 32));     // combine wave halves
      float newMax = fmaxf(rowMax, cmax);
      float corr   = __expf(rowMax - newMax);
      #pragma unroll
      for (int t = 0; t < 16; ++t) dacc[t] *= corr;
      rowSum *= corr;
      rowMax  = newMax;
      // probs, packed directly into the attn^T B-operand (lane-local)
      union { v16bf v; unsigned u[8]; } pb;
      float psum = 0.0f;
      #pragma unroll
      for (int r = 0; r < 4; ++r) {
        float p00 = __expf(s0[2 * r]     - newMax);
        float p01 = __expf(s0[2 * r + 1] - newMax);
        float p10 = __expf(s1[2 * r]     - newMax);
        float p11 = __expf(s1[2 * r + 1] - newMax);
        psum += p00 + p01 + p10 + p11;
        pb.u[r]     = pack2(p00, p01);
        pb.u[4 + r] = pack2(p10, p11);
      }
      rowSum += psum;
      // GEMM2: delta^T += V^T(chunk) x attn^T(chunk)
      #pragma unroll
      for (int dt = 0; dt < 16; ++dt) {
        const v16bf* va = (const v16bf*)(smem + LDS_VF + ((mc * 16 + dt) << 10) + lane * 32);
        dacc[dt] = __builtin_amdgcn_wmma_f32_16x16x32_bf16(false, *va, false, pb.v,
                                                           (short)0, dacc[dt], false, false);
      }
    }

    // softmax normalization (halves hold disjoint m-partials of the row sum)
    float totSum = rowSum + __shfl_xor(rowSum, 16, 32);
    float invSum = (totSum > 0.0f) ? (1.0f / totSum) : 0.0f;

    // gate = sigmoid(alpha * (y . delta)/D + bias)
    float dotp = 0.0f;
    #pragma unroll
    for (int t = 0; t < 16; ++t) {
      #pragma unroll
      for (int r = 0; r < 8; ++r) dotp += yT[t][r] * dacc[t][r];
    }
    dotp *= invSum;
    dotp += __shfl_xor(dotp, 16, 32);
    dotp *= (1.0f / (float)D_);
    float gate = 1.0f / (1.0f + __expf(-(alpha * dotp + gbias)));

    float gi = gate * invSum;
    #pragma unroll
    for (int t = 0; t < 16; ++t) {
      #pragma unroll
      for (int r = 0; r < 8; ++r) yT[t][r] += gi * dacc[t][r];
    }
  }

  // out += y_b   (out was pre-initialized to -4*seed; each lane owns row myRow)
  float* orow = out + ((size_t)bs * N_ + myRow) * D_;
  #pragma unroll
  for (int t = 0; t < 16; ++t) {
    int dbase = t * 16 + 8 * half;
    #pragma unroll
    for (int r = 0; r < 8; ++r)
      unsafeAtomicAdd(orow + dbase + r, yT[t][r]);
  }
}

extern "C" void kernel_launch(void* const* d_in, const int* in_sizes, int n_in,
                              void* d_out, int out_size, void* d_ws, size_t ws_size,
                              hipStream_t stream) {
  const float* seed = (const float*)d_in[0];
  const float* emK  = (const float*)d_in[1];
  const float* emV  = (const float*)d_in[2];
  const float* emS  = (const float*)d_in[3];
  const float* gA   = (const float*)d_in[4];
  const float* gB   = (const float*)d_in[5];
  const float* rT   = (const float*)d_in[6];
  float* out = (float*)d_out;

  (void)in_sizes; (void)n_in; (void)d_ws; (void)ws_size; (void)out_size;

  hipFuncSetAttribute(reinterpret_cast<const void*>(ep_mem_kernel),
                      hipFuncAttributeMaxDynamicSharedMemorySize, LDS_TOTAL);

  const int total = BS_ * N_ * D_;
  ep_init_out<<<(total + 255) / 256, 256, 0, stream>>>(seed, out, total);

  dim3 grid(NT, B_, BS_);
  ep_mem_kernel<<<grid, 256, LDS_TOTAL, stream>>>(seed, emK, emV, emS, gA, gB, rT, out);
}
